// CausalSelfAttention_2224793059575
// MI455X (gfx1250) — matrix operand order repaired and checked by reading, not compiled, not run on hardware
//
#include <hip/hip_runtime.h>

#define T_SEQ   2048
#define D_MODEL 1024
#define N_HEADS 8
#define HEAD_DIM 128
#define GATE_IN 12
#define ATTN_SCALE 0.12f
#define EPS_RMS 1e-6f

typedef __attribute__((ext_vector_type(16))) __bf16 v16bf;
typedef __attribute__((ext_vector_type(8)))  __bf16 bf8v;
typedef __attribute__((ext_vector_type(8)))  float  v8f;
typedef int v4i __attribute__((vector_size(16)));   // matches builtin param type

static __device__ __forceinline__ __bf16 f2bf(float f) { return (__bf16)f; }

static __device__ __forceinline__ v16bf cat8(bf8v lo, bf8v hi) {
  return __builtin_shufflevector(lo, hi, 0,1,2,3,4,5,6,7,8,9,10,11,12,13,14,15);
}

static __device__ __forceinline__ v8f wmma_bf16(v16bf a, v16bf b, v8f c) {
  return __builtin_amdgcn_wmma_f32_16x16x32_bf16(false, a, false, b, (short)0, c, false, false);
}

// ---- CDNA5 async global->LDS copies (ASYNCcnt path), guarded -------------
#if defined(__HIP_DEVICE_COMPILE__) && __has_builtin(__builtin_amdgcn_global_load_async_to_lds_b128)
#define HAVE_ASYNC_LDS 1
#else
#define HAVE_ASYNC_LDS 0
#endif

#if HAVE_ASYNC_LDS
#define AS3 __attribute__((address_space(3)))
#define AS1 __attribute__((address_space(1)))
static __device__ __forceinline__ AS3 v4i* to_lds(void* p) {
  // matches LLVM flat->local addrspacecast (trunc to 32-bit LDS offset)
  return (AS3 v4i*)(unsigned int)(unsigned long long)p;
}
static __device__ __forceinline__ AS1 v4i* to_glb(const void* p) {
  return (AS1 v4i*)(unsigned long long)p;
}
static __device__ __forceinline__ void wait_async0() {
#if __has_builtin(__builtin_amdgcn_s_wait_asynccnt)
  __builtin_amdgcn_s_wait_asynccnt(0);
#else
  asm volatile("s_wait_asynccnt 0x0" ::: "memory");
#endif
}
#endif

// copy 32 bytes (16 bf16) global -> LDS
static __device__ __forceinline__ void g2l_32B(const __bf16* g, __bf16* l) {
#if HAVE_ASYNC_LDS
  __builtin_amdgcn_global_load_async_to_lds_b128(to_glb(g), to_lds(l), 0, 0);
  __builtin_amdgcn_global_load_async_to_lds_b128(to_glb(g), to_lds(l), 16, 0);
#else
  *(uint4*)l       = *(const uint4*)g;
  *(uint4*)(l + 8) = *(const uint4*)(g + 8);
#endif
}

// ---------------------------------------------------------------------------
// f32 -> bf16 bulk convert (8 elems/thread)
// ---------------------------------------------------------------------------
__global__ __launch_bounds__(256)
void cvt_bf16_kernel(const float* __restrict__ src, unsigned short* __restrict__ dst, int n) {
  int i = (blockIdx.x * 256 + threadIdx.x) * 8;
  if (i >= n) return;
  float4 a = *(const float4*)(src + i);
  float4 b = *(const float4*)(src + i + 4);
  __bf16* d = (__bf16*)dst + i;
  d[0]=f2bf(a.x); d[1]=f2bf(a.y); d[2]=f2bf(a.z); d[3]=f2bf(a.w);
  d[4]=f2bf(b.x); d[5]=f2bf(b.y); d[6]=f2bf(b.z); d[7]=f2bf(b.w);
}

// ---------------------------------------------------------------------------
// GEMM: C[M,N] = A[M,K] * B[N,K]^T  (A,B bf16 row-major, C f32)
// Block: 256 threads (8 waves). Block tile 128x128, wave tile 32x64.
// Double-buffered LDS filled by async global->LDS copies.
// ---------------------------------------------------------------------------
__global__ __launch_bounds__(256)
void gemm_bf16_kernel(const unsigned short* __restrict__ A_,
                      const unsigned short* __restrict__ B_,
                      float* __restrict__ C, int M, int N, int K) {
  const __bf16* A = (const __bf16*)A_;
  const __bf16* B = (const __bf16*)B_;

  __shared__ __bf16 As[2][128][40];
  __shared__ __bf16 Bs[2][128][40];

  const int tid  = threadIdx.x;
  const int lane = tid & 31;
  const int wave = tid >> 5;
  const int lm   = lane & 15;
  const int lg   = lane >> 4;
  const int wm   = wave & 3;   // 0..3 : 32-row slab
  const int wn   = wave >> 2;  // 0..1 : 64-col slab
  const int m0   = blockIdx.y * 128;
  const int n0   = blockIdx.x * 128;

  const int crow = tid >> 1;          // 0..127
  const int ccol = (tid & 1) << 4;    // 0 or 16

  v8f acc[2][4];
  #pragma unroll
  for (int i = 0; i < 2; ++i)
    #pragma unroll
    for (int j = 0; j < 4; ++j) acc[i][j] = (v8f){};

  // stage K-tile kk into buffer p (each thread: 2x32B async copies)
  auto stage = [&](int p, int kk) {
    g2l_32B(A + (size_t)(m0 + crow) * K + kk + ccol, &As[p][crow][ccol]);
    g2l_32B(B + (size_t)(n0 + crow) * K + kk + ccol, &Bs[p][crow][ccol]);
  };

  stage(0, 0);
  int p = 0;
  for (int kk = 0; kk < K; kk += 32) {
#if HAVE_ASYNC_LDS
    wait_async0();
#endif
    __syncthreads();
    if (kk + 32 < K) stage(p ^ 1, kk + 32);

    v16bf af[2];
    #pragma unroll
    for (int i = 0; i < 2; ++i) {
      const __bf16* ar = &As[p][wm*32 + i*16 + lm][0];
      af[i] = cat8(*(const bf8v*)(ar + lg*16), *(const bf8v*)(ar + lg*16 + 8));
    }
    v16bf bfr[4];
    #pragma unroll
    for (int j = 0; j < 4; ++j) {
      const __bf16* br = &Bs[p][wn*64 + j*16 + lm][lg*16];
      bfr[j] = cat8(*(const bf8v*)br, *(const bf8v*)(br + 8));
    }
    #pragma unroll
    for (int i = 0; i < 2; ++i)
      #pragma unroll
      for (int j = 0; j < 4; ++j)
        acc[i][j] = wmma_bf16(af[i], bfr[j], acc[i][j]);

    p ^= 1;
  }

  #pragma unroll
  for (int i = 0; i < 2; ++i)
    #pragma unroll
    for (int j = 0; j < 4; ++j)
      #pragma unroll
      for (int r = 0; r < 8; ++r) {
        int row = m0 + wm*32 + i*16 + r + 8*lg;
        C[(size_t)row * N + n0 + wn*64 + j*16 + lm] = acc[i][j][r];
      }
}

// ---------------------------------------------------------------------------
// Prep: RMSNorm + rotary on q,k; v = l0*v + l1*ve; sigmoid gate.
// Q,K -> bf16 [h][t][hd]; V -> bf16 transposed [h][hd][t]; gate f32 [t][h].
// ---------------------------------------------------------------------------
__global__ __launch_bounds__(128)
void prep_kernel(const float* __restrict__ qkv, const float* __restrict__ x,
                 const float* __restrict__ ve, const float* __restrict__ lam,
                 const float* __restrict__ cosb, const float* __restrict__ sinb,
                 const float* __restrict__ gate_w,
                 unsigned short* __restrict__ Qb_, unsigned short* __restrict__ Kb_,
                 unsigned short* __restrict__ Vt_, float* __restrict__ gate) {
  __bf16* Qb = (__bf16*)Qb_; __bf16* Kb = (__bf16*)Kb_; __bf16* Vt = (__bf16*)Vt_;
  const int t = blockIdx.x, h = blockIdx.y, i = threadIdx.x;

  __shared__ float red[128];
  __shared__ float qn[128];
  __shared__ float kn[128];

  const size_t rowbase = (size_t)t * (3 * D_MODEL) + h * HEAD_DIM;
  float qv = qkv[rowbase + i];
  float kv = qkv[rowbase + D_MODEL + i];
  float vv = qkv[rowbase + 2 * D_MODEL + i];

  red[i] = qv * qv; __syncthreads();
  #pragma unroll
  for (int s = 64; s > 0; s >>= 1) { if (i < s) red[i] += red[i + s]; __syncthreads(); }
  float qr = rsqrtf(red[0] * (1.0f / HEAD_DIM) + EPS_RMS);
  __syncthreads();
  red[i] = kv * kv; __syncthreads();
  #pragma unroll
  for (int s = 64; s > 0; s >>= 1) { if (i < s) red[i] += red[i + s]; __syncthreads(); }
  float kr = rsqrtf(red[0] * (1.0f / HEAD_DIM) + EPS_RMS);
  __syncthreads();

  qn[i] = qv * qr; kn[i] = kv * kr; __syncthreads();

  float qo, ko;
  if (i < 64) {
    float c = cosb[t * 64 + i], s = sinb[t * 64 + i];
    qo =  qn[i] * c + qn[i + 64] * s;
    ko =  kn[i] * c + kn[i + 64] * s;
  } else {
    int j = i - 64;
    float c = cosb[t * 64 + j], s = sinb[t * 64 + j];
    qo = -qn[j] * s + qn[i] * c;
    ko = -kn[j] * s + kn[i] * c;
  }

  const size_t qk_idx = ((size_t)h * T_SEQ + t) * HEAD_DIM + i;
  Qb[qk_idx] = f2bf(qo);
  Kb[qk_idx] = f2bf(ko);

  float vb = lam[0] * vv + lam[1] * ve[((size_t)t * N_HEADS + h) * HEAD_DIM + i];
  Vt[((size_t)h * HEAD_DIM + i) * T_SEQ + t] = f2bf(vb);

  if (i == 0) {
    float acc = 0.f;
    #pragma unroll
    for (int j = 0; j < GATE_IN; ++j) acc += x[(size_t)t * D_MODEL + j] * gate_w[h * GATE_IN + j];
    gate[t * N_HEADS + h] = 1.0f / (1.0f + __expf(-acc));
  }
}

// ---------------------------------------------------------------------------
// Flash attention: 1 wave per (head, 16-query tile). KV tiles of 32 keys.
// S = Q K^T (8 wmma), online softmax, O += P V (8 wmma). Writes gated bf16 Y.
// ---------------------------------------------------------------------------
__global__ __launch_bounds__(32)
void attn_kernel(const unsigned short* __restrict__ Qb_,
                 const unsigned short* __restrict__ Kb_,
                 const unsigned short* __restrict__ Vt_,
                 const float* __restrict__ gate, unsigned short* __restrict__ Yb_) {
  const __bf16* Qb = (const __bf16*)Qb_;
  const __bf16* Kb = (const __bf16*)Kb_;
  const __bf16* Vt = (const __bf16*)Vt_;
  __bf16* Yb = (__bf16*)Yb_;

  __shared__ __bf16 Ps[16][40];

  const int h  = blockIdx.y;
  const int m0 = blockIdx.x * 16;
  const int lane = threadIdx.x;
  const int lm = lane & 15;
  const int lg = lane >> 4;

  const __bf16* Qh = Qb + (size_t)h * T_SEQ * HEAD_DIM;
  const __bf16* Kh = Kb + (size_t)h * T_SEQ * HEAD_DIM;
  const __bf16* Vh = Vt + (size_t)h * HEAD_DIM * T_SEQ;

  v16bf qf[4];
  #pragma unroll
  for (int c = 0; c < 4; ++c) {
    const __bf16* row = Qh + (size_t)(m0 + lm) * HEAD_DIM + 32 * c;
    qf[c] = cat8(*(const bf8v*)(row + lg * 16), *(const bf8v*)(row + lg * 16 + 8));
  }

  v8f o[8];
  #pragma unroll
  for (int c = 0; c < 8; ++c) o[c] = (v8f){};
  float mrow[8], lrow[8];
  #pragma unroll
  for (int r = 0; r < 8; ++r) { mrow[r] = -1e30f; lrow[r] = 0.f; }

  const int kend = m0 + 16;
  for (int kt = 0; kt < kend; kt += 32) {
    v8f s0 = (v8f){}, s1 = (v8f){};
    #pragma unroll
    for (int c = 0; c < 4; ++c) {
      const __bf16* k0 = Kh + (size_t)(kt + lm) * HEAD_DIM + 32 * c + lg * 16;
      const __bf16* k1 = Kh + (size_t)(kt + 16 + lm) * HEAD_DIM + 32 * c + lg * 16;
      s0 = wmma_bf16(qf[c], cat8(*(const bf8v*)k0, *(const bf8v*)(k0 + 8)), s0);
      s1 = wmma_bf16(qf[c], cat8(*(const bf8v*)k1, *(const bf8v*)(k1 + 8)), s1);
    }

    __syncthreads();

    #pragma unroll
    for (int r = 0; r < 8; ++r) {
      const int row = r + 8 * lg;
      const int tq  = m0 + row;
      float v0 = s0[r] * ATTN_SCALE;
      float v1 = s1[r] * ATTN_SCALE;
      if (kt + lm      > tq) v0 = -1e30f;
      if (kt + 16 + lm > tq) v1 = -1e30f;
      float mx = fmaxf(v0, v1);
      mx = fmaxf(mx, __shfl_xor(mx, 1));
      mx = fmaxf(mx, __shfl_xor(mx, 2));
      mx = fmaxf(mx, __shfl_xor(mx, 4));
      mx = fmaxf(mx, __shfl_xor(mx, 8));
      float mnew = fmaxf(mrow[r], mx);
      float p0 = __expf(v0 - mnew);
      float p1 = __expf(v1 - mnew);
      float rs = p0 + p1;
      rs += __shfl_xor(rs, 1);
      rs += __shfl_xor(rs, 2);
      rs += __shfl_xor(rs, 4);
      rs += __shfl_xor(rs, 8);
      float corr = __expf(mrow[r] - mnew);
      lrow[r] = lrow[r] * corr + rs;
      mrow[r] = mnew;
      #pragma unroll
      for (int c = 0; c < 8; ++c) o[c][r] *= corr;
      Ps[row][lm]      = f2bf(p0);
      Ps[row][16 + lm] = f2bf(p1);
    }
    __syncthreads();

    bf8v plo = *(const bf8v*)(&Ps[lm][lg * 16]);
    bf8v phi = *(const bf8v*)(&Ps[lm][lg * 16 + 8]);
    v16bf pf = cat8(plo, phi);

    #pragma unroll
    for (int c = 0; c < 8; ++c) {
      const __bf16* vp = Vh + (size_t)(16 * c + lm) * T_SEQ + kt + lg * 16;
      o[c] = wmma_bf16(pf, cat8(*(const bf8v*)vp, *(const bf8v*)(vp + 8)), o[c]);
    }
  }

  #pragma unroll
  for (int r = 0; r < 8; ++r) {
    const int tq = m0 + r + 8 * lg;
    const float f = (1.0f / lrow[r]) * gate[tq * N_HEADS + h];
    #pragma unroll
    for (int c = 0; c < 8; ++c) {
      Yb[(size_t)tq * D_MODEL + h * HEAD_DIM + 16 * c + lm] = f2bf(o[c][r] * f);
    }
  }
}

// ---------------------------------------------------------------------------
extern "C" void kernel_launch(void* const* d_in, const int* in_sizes, int n_in,
                              void* d_out, int out_size, void* d_ws, size_t ws_size,
                              hipStream_t stream) {
  const float* x      = (const float*)d_in[0];
  const float* qkvo_w = (const float*)d_in[1];
  const float* gate_w = (const float*)d_in[2];
  const float* ve     = (const float*)d_in[3];
  const float* lam    = (const float*)d_in[4];
  const float* cosb   = (const float*)d_in[5];
  const float* sinb   = (const float*)d_in[6];
  (void)in_sizes; (void)n_in; (void)out_size; (void)ws_size;

  char* ws = (char*)d_ws;
  float* qkv = (float*)ws;                                   // T*3D f32
  ws += (size_t)T_SEQ * 3 * D_MODEL * sizeof(float);
  unsigned short* Qb = (unsigned short*)ws;                  // H*T*HD bf16
  ws += (size_t)N_HEADS * T_SEQ * HEAD_DIM * 2;
  unsigned short* Kb = (unsigned short*)ws;
  ws += (size_t)N_HEADS * T_SEQ * HEAD_DIM * 2;
  unsigned short* Vt = (unsigned short*)ws;                  // [h][hd][t] bf16
  ws += (size_t)N_HEADS * HEAD_DIM * T_SEQ * 2;
  float* gate = (float*)ws;                                  // T*H f32
  ws += (size_t)T_SEQ * N_HEADS * sizeof(float);
  unsigned short* Xb = (unsigned short*)ws;                  // T*D bf16
  ws += (size_t)T_SEQ * D_MODEL * 2;
  unsigned short* Wb = (unsigned short*)ws;                  // 4*D*D bf16
  ws += (size_t)4 * D_MODEL * D_MODEL * 2;
  unsigned short* Yb = (unsigned short*)ws;                  // T*D bf16
  ws += (size_t)T_SEQ * D_MODEL * 2;

  // 0) one-time f32 -> bf16 conversion of activations and all weights
  {
    int nx = T_SEQ * D_MODEL;
    cvt_bf16_kernel<<<nx / 8 / 256, 256, 0, stream>>>(x, Xb, nx);
    int nw = 4 * D_MODEL * D_MODEL;
    cvt_bf16_kernel<<<nw / 8 / 256, 256, 0, stream>>>(qkvo_w, Wb, nw);
  }

  // 1) qkv = x @ w_qkv^T      (M=2048, N=3072, K=1024)
  gemm_bf16_kernel<<<dim3(3 * D_MODEL / 128, T_SEQ / 128), 256, 0, stream>>>(
      Xb, Wb, qkv, T_SEQ, 3 * D_MODEL, D_MODEL);

  // 2) rmsnorm + rotary + v-blend + gate
  prep_kernel<<<dim3(T_SEQ, N_HEADS), 128, 0, stream>>>(
      qkv, x, ve, lam, cosb, sinb, gate_w, Qb, Kb, Vt, gate);

  // 3) flash attention (gated), bf16 output
  attn_kernel<<<dim3(T_SEQ / 16, N_HEADS), 32, 0, stream>>>(Qb, Kb, Vt, gate, Yb);

  // 4) out = Y @ w_o^T        (M=2048, N=1024, K=1024)
  gemm_bf16_kernel<<<dim3(D_MODEL / 128, T_SEQ / 128), 256, 0, stream>>>(
      Yb, Wb + (size_t)3 * D_MODEL * D_MODEL, (float*)d_out, T_SEQ, D_MODEL, D_MODEL);
}
